// MultiHeadedAttention_Carbon_80290118631799
// MI455X (gfx1250) — matrix operand order repaired and checked by reading, not compiled, not run on hardware
//
#include <hip/hip_runtime.h>
#include <math.h>

// ---------------------------------------------------------------------------
// Problem constants (from reference)
// ---------------------------------------------------------------------------
constexpr int B_      = 4;
constexpr int S_      = 2048;
constexpr int HID     = 1024;
constexpr int FAC     = 256;
constexpr int NH      = 16;
constexpr int DK      = 64;
constexpr int BH      = B_ * NH;          // 64
constexpr int MROWS   = B_ * S_;          // 8192

typedef __attribute__((ext_vector_type(16))) _Float16 v16h;
typedef __attribute__((ext_vector_type(8)))  _Float16 v8h;
typedef __attribute__((ext_vector_type(8)))  float    v8f;

static __device__ __forceinline__ v16h ldfrag(const _Float16* tile, int lane) {
    return *reinterpret_cast<const v16h*>(tile + lane * 16);
}
static __device__ __forceinline__ v8f wmma(v16h a, v16h b, v8f c) {
    return __builtin_amdgcn_wmma_f32_16x16x32_f16(false, a, false, b, (short)0, c, false, false);
}

// DPP16 butterfly max across the 16 lanes of a half-wave (row group).
// Stages: quad_perm(1,0,3,2), quad_perm(2,3,0,1), row_half_mirror, row_mirror.
// dpp_ctrl must be an immediate -> template parameter.
template <int CTRL>
static __device__ __forceinline__ float dpp_max_step(float x) {
    int t = __builtin_amdgcn_update_dpp(0, __float_as_int(x), CTRL, 0xf, 0xf, true);
    return fmaxf(x, __int_as_float(t));
}
static __device__ __forceinline__ float rowmax16(float x) {
    x = dpp_max_step<0xB1>(x);   // quad_perm(1,0,3,2)
    x = dpp_max_step<0x4E>(x);   // quad_perm(2,3,0,1)
    x = dpp_max_step<0x141>(x);  // row_half_mirror
    x = dpp_max_step<0x140>(x);  // row_mirror
    return x;
}

// ---------------------------------------------------------------------------
// Pack row-major [M,K] (float or f16) into WMMA A-fragment layout.
// Tile (mt,kt) -> 512 halves at (mt*Ktiles+kt)*512; lane l holds row mt*16+(l&15),
// K offsets kt*32 + (l>>4)*16 + {0..15}.  (contiguous k order; B packs use the same)
// ---------------------------------------------------------------------------
template <typename T>
__global__ void pack_a_kernel(const T* __restrict__ src, _Float16* __restrict__ dst,
                              int M, int K) {
    int wid  = (blockIdx.x * blockDim.x + threadIdx.x) >> 5;
    int lane = threadIdx.x & 31;
    int ktiles = K >> 5;
    int mt = wid / ktiles, kt = wid % ktiles;
    if (mt >= (M >> 4)) return;
    int row = mt * 16 + (lane & 15);
    int kb  = kt * 32 + (lane >> 4) * 16;
    const T* s = src + (size_t)row * K + kb;
    v16h v;
#pragma unroll
    for (int e = 0; e < 8; ++e) {
        v[e]     = (_Float16)s[e];
        v[e + 8] = (_Float16)s[8 + e];
    }
    *reinterpret_cast<v16h*>(dst + (size_t)wid * 512 + lane * 16) = v;
}

// ---------------------------------------------------------------------------
// Pack fp32 weight [K,N] (row-major) into WMMA B-fragment layout.
// Tile (kt,nt) -> 512 halves at (kt*Ntiles+nt)*512; lane l holds col nt*16+(l&15),
// K offsets kt*32 + (l>>4)*16 + e.  (contiguous k order, same as A packs)
// ---------------------------------------------------------------------------
__global__ void pack_b_kernel(const float* __restrict__ src, _Float16* __restrict__ dst,
                              int K, int N) {
    int wid  = (blockIdx.x * blockDim.x + threadIdx.x) >> 5;
    int lane = threadIdx.x & 31;
    int ntiles = N >> 4;
    int kt = wid / ntiles, nt = wid % ntiles;
    if (kt >= (K >> 5)) return;
    int n  = nt * 16 + (lane & 15);
    int kb = kt * 32 + (lane >> 4) * 16;
    v16h v;
#pragma unroll
    for (int e = 0; e < 16; ++e) v[e] = (_Float16)src[(size_t)(kb + e) * N + n];
    *reinterpret_cast<v16h*>(dst + (size_t)wid * 512 + lane * 16) = v;
}

// ---------------------------------------------------------------------------
// Pack K^T fragments for attention: kh is f16 [BH][S][DK] row-major.
// B-matrix is K^T (DK x S).  lane l: s = nt*16+(l&15), d = kt*32+(l>>4)*16 + e
// -> 16 contiguous halves of kh -> single 32B load.
// dst tile order: ((bh*2 + kt)*128 + nt)*512
// ---------------------------------------------------------------------------
__global__ void pack_kt_kernel(const _Float16* __restrict__ kh, _Float16* __restrict__ dst) {
    int wid  = (blockIdx.x * blockDim.x + threadIdx.x) >> 5;
    int lane = threadIdx.x & 31;
    int bh = wid >> 8, rem = wid & 255;
    int kt = rem >> 7, nt = rem & 127;
    int s  = nt * 16 + (lane & 15);
    int d  = kt * 32 + (lane >> 4) * 16;
    v16h v = *reinterpret_cast<const v16h*>(kh + ((size_t)bh * S_ + s) * DK + d);
    *reinterpret_cast<v16h*>(dst + (size_t)wid * 512 + lane * 16) = v;
}

// ---------------------------------------------------------------------------
// Pack V fragments: vh f16 [BH][S][DK];  B-matrix is V (S x DK).
// lane l: d = nt*16+(l&15), s = kt*32+(l>>4)*16 + e  (stride-DK gather).
// dst tile order: ((bh*64 + kt)*4 + nt)*512
// ---------------------------------------------------------------------------
__global__ void pack_v_kernel(const _Float16* __restrict__ vh, _Float16* __restrict__ dst) {
    int wid  = (blockIdx.x * blockDim.x + threadIdx.x) >> 5;
    int lane = threadIdx.x & 31;
    int bh = wid >> 8, rem = wid & 255;
    int kt = rem >> 2, nt = rem & 3;
    int d  = nt * 16 + (lane & 15);
    int sb = kt * 32 + (lane >> 4) * 16;
    v16h v;
#pragma unroll
    for (int e = 0; e < 16; ++e) v[e] = vh[((size_t)bh * S_ + sb + e) * DK + d];
    *reinterpret_cast<v16h*>(dst + (size_t)wid * 512 + lane * 16) = v;
}

// ---------------------------------------------------------------------------
// WMMA GEMM:  D = A_pack[M,K] x B_pack[K,N] + bias, with fused epilogues.
// Block: 128 threads (4 waves).  Block tile 128(M) x 64(N); wave = 32x64.
// EPI 0: leaky-relu(0.2), write row-major f16 [M,N]
// EPI 1: write f16 head-split [B][H][S][DK]   (m=b*S+s, n=h*DK+d)
// EPI 2: write row-major f32 [M,N]  (final output)
// ---------------------------------------------------------------------------
template <int EPI>
__global__ void gemm_wmma_kernel(const _Float16* __restrict__ A,
                                 const _Float16* __restrict__ Bp,
                                 const float* __restrict__ bias,
                                 void* __restrict__ out,
                                 int M, int N, int K) {
    const int lane = threadIdx.x & 31;
    const int w    = threadIdx.x >> 5;
    const int ktiles = K >> 5;
    const int ntiles = N >> 4;
    const int mt0 = blockIdx.y * 8 + w * 2;
    const int nt0 = blockIdx.x * 4;

    v8f acc[2][4];
#pragma unroll
    for (int i = 0; i < 2; ++i)
#pragma unroll
        for (int j = 0; j < 4; ++j) acc[i][j] = v8f{0.f,0.f,0.f,0.f,0.f,0.f,0.f,0.f};

    for (int kt = 0; kt < ktiles; ++kt) {
        v16h a0 = ldfrag(A + (size_t)(mt0 * ktiles + kt) * 512, lane);
        v16h a1 = ldfrag(A + (size_t)((mt0 + 1) * ktiles + kt) * 512, lane);
        if (kt + 1 < ktiles)
            __builtin_prefetch(A + (size_t)(mt0 * ktiles + kt + 1) * 512 + lane * 16, 0, 3);
#pragma unroll
        for (int nt = 0; nt < 4; ++nt) {
            v16h b = ldfrag(Bp + (size_t)(kt * ntiles + nt0 + nt) * 512, lane);
            acc[0][nt] = wmma(a0, b, acc[0][nt]);
            acc[1][nt] = wmma(a1, b, acc[1][nt]);
        }
    }

#pragma unroll
    for (int i = 0; i < 2; ++i) {
        int rbase = (mt0 + i) * 16 + (lane >> 4) * 8;
#pragma unroll
        for (int nt = 0; nt < 4; ++nt) {
            int col = (nt0 + nt) * 16 + (lane & 15);
            float bz = bias[col];
#pragma unroll
            for (int r = 0; r < 8; ++r) {
                float y = acc[i][nt][r] + bz;
                if (EPI == 0) {
                    y = (y > 0.f) ? y : 0.2f * y;
                    ((_Float16*)out)[(size_t)(rbase + r) * N + col] = (_Float16)y;
                } else if (EPI == 1) {
                    int m = rbase + r;
                    int b = m >> 11, s = m & (S_ - 1);
                    int h = col >> 6, d = col & (DK - 1);
                    ((_Float16*)out)[(((size_t)(b * NH + h)) * S_ + s) * DK + d] = (_Float16)y;
                } else {
                    ((float*)out)[(size_t)(rbase + r) * N + col] = y;
                }
            }
        }
    }
}

// ---------------------------------------------------------------------------
// Flash attention per (bh, 64-query block).  Block = 128 threads = 4 waves;
// each wave owns one 16-row query tile.  Online softmax over 64 key-steps of
// 32 keys.  Row max: DPP16 butterfly (pure VALU, no LDS).  Row sum: extra
// WMMA against an all-ones B fragment (P @ ones replicates row sums into
// every column/lane, in the same C layout as the accumulators).
// P is re-laid-out C->A fragment via per-wave LDS staging tile.
// Output: cv f16 merged-head row-major [B*S][HID].
// ---------------------------------------------------------------------------
__global__ void flash_attn_kernel(const _Float16* __restrict__ Qp,
                                  const _Float16* __restrict__ KTp,
                                  const _Float16* __restrict__ Vp,
                                  const int* __restrict__ mask,
                                  _Float16* __restrict__ cv) {
    __shared__ float maskb[S_];
    __shared__ __align__(32) _Float16 pstage[4][512];   // per-wave 16x32 tile

    const int bh = blockIdx.y;
    const int b  = bh >> 4;
    const int h  = bh & 15;

    for (int i = threadIdx.x; i < S_; i += blockDim.x)
        maskb[i] = (mask[b * S_ + i] == 0) ? -1e20f : 0.0f;
    __syncthreads();

    const int lane = threadIdx.x & 31;
    const int w    = threadIdx.x >> 5;
    const int qt   = blockIdx.x * 4 + w;          // query tile within bh (0..127)
    const int colg = lane >> 4;                   // row-group (0/1)
    const int col  = lane & 15;

    const _Float16* Qt  = Qp  + (size_t)(bh * 128 + qt) * 2 * 512;
    const _Float16* KTb = KTp + (size_t)bh * 2 * 128 * 512;
    const _Float16* Vb  = Vp  + (size_t)bh * 64 * 4 * 512;
    _Float16* myp = &pstage[w][0];

    v16h q0 = ldfrag(Qt, lane);
    v16h q1 = ldfrag(Qt + 512, lane);

    v16h ones;
#pragma unroll
    for (int e = 0; e < 16; ++e) ones[e] = (_Float16)1.0f;

    float mrow[8];
    v8f acc[4], asum;
#pragma unroll
    for (int r = 0; r < 8; ++r) mrow[r] = -3.0e38f;
    asum = v8f{0.f,0.f,0.f,0.f,0.f,0.f,0.f,0.f};
#pragma unroll
    for (int nt = 0; nt < 4; ++nt) acc[nt] = v8f{0.f,0.f,0.f,0.f,0.f,0.f,0.f,0.f};

    for (int ks = 0; ks < 64; ++ks) {             // 32 keys per step
        int j0 = ks * 2, j1 = j0 + 1;
        v8f s0 = v8f{0.f,0.f,0.f,0.f,0.f,0.f,0.f,0.f};
        v8f s1 = s0;
        s0 = wmma(q0, ldfrag(KTb + (size_t)(0 * 128 + j0) * 512, lane), s0);
        s0 = wmma(q1, ldfrag(KTb + (size_t)(1 * 128 + j0) * 512, lane), s0);
        s1 = wmma(q0, ldfrag(KTb + (size_t)(0 * 128 + j1) * 512, lane), s1);
        s1 = wmma(q1, ldfrag(KTb + (size_t)(1 * 128 + j1) * 512, lane), s1);

        if (ks + 1 < 64) {   // keep the L2-resident K/V stream ahead of WMMA issue
            __builtin_prefetch(KTb + (size_t)(j0 + 2) * 512 + lane * 16, 0, 3);
            __builtin_prefetch(KTb + (size_t)(128 + j0 + 2) * 512 + lane * 16, 0, 3);
            __builtin_prefetch(Vb + (size_t)((ks + 1) * 4) * 512 + lane * 16, 0, 3);
        }

        float mb0 = maskb[j0 * 16 + col];
        float mb1 = maskb[j1 * 16 + col];

        float corr[8];
#pragma unroll
        for (int r = 0; r < 8; ++r) {
            float sm0 = s0[r] * 0.125f + mb0;     // 1/sqrt(64)
            float sm1 = s1[r] * 0.125f + mb1;
            float t = rowmax16(fmaxf(sm0, sm1));  // DPP butterfly, no DS ops
            float mnew = fmaxf(mrow[r], t);
            corr[r] = __expf(mrow[r] - mnew);
            mrow[r] = mnew;
            float p0 = __expf(sm0 - mnew);
            float p1 = __expf(sm1 - mnew);
            int row = colg * 8 + r;
            myp[row * 32 + col]      = (_Float16)p0;
            myp[row * 32 + 16 + col] = (_Float16)p1;
        }
#pragma unroll
        for (int nt = 0; nt < 4; ++nt)
#pragma unroll
            for (int r = 0; r < 8; ++r) acc[nt][r] *= corr[r];
#pragma unroll
        for (int r = 0; r < 8; ++r) asum[r] *= corr[r];

        // P tile C-layout -> A-layout via LDS (same-wave DS ops are in order)
        v8h lo = *reinterpret_cast<const v8h*>(myp + (lane & 15) * 32 + colg * 16);
        v8h hi = *reinterpret_cast<const v8h*>(myp + (lane & 15) * 32 + colg * 16 + 8);
        v16h pf;
#pragma unroll
        for (int e = 0; e < 8; ++e) { pf[e] = lo[e]; pf[e + 8] = hi[e]; }

        // Row sums of P via WMMA: P @ ones -> every lane gets its rows' sums.
        asum = wmma(pf, ones, asum);

#pragma unroll
        for (int nt = 0; nt < 4; ++nt) {
            v16h vf = ldfrag(Vb + (size_t)(ks * 4 + nt) * 512, lane);
            acc[nt] = wmma(pf, vf, acc[nt]);
        }
    }

    float inv[8];
#pragma unroll
    for (int r = 0; r < 8; ++r) inv[r] = 1.0f / asum[r];
#pragma unroll
    for (int nt = 0; nt < 4; ++nt) {
#pragma unroll
        for (int r = 0; r < 8; ++r) {
            int s = qt * 16 + colg * 8 + r;
            int c = h * DK + nt * 16 + col;
            cv[((size_t)b * S_ + s) * HID + c] = (_Float16)(acc[nt][r] * inv[r]);
        }
    }
}

// ---------------------------------------------------------------------------
// Host: orchestrate the whole pipeline.
// ---------------------------------------------------------------------------
extern "C" void kernel_launch(void* const* d_in, const int* in_sizes, int n_in,
                              void* d_out, int out_size, void* d_ws, size_t ws_size,
                              hipStream_t stream) {
    const float* x_q   = (const float*)d_in[0];
    const float* x_k   = (const float*)d_in[1];
    const float* x_v   = (const float*)d_in[2];
    const int*   mask  = (const int*)d_in[3];
    const float* Wp[4] = {(const float*)d_in[4],  (const float*)d_in[8],
                          (const float*)d_in[12], (const float*)d_in[16]};
    const float* bp[4] = {(const float*)d_in[5],  (const float*)d_in[9],
                          (const float*)d_in[13], (const float*)d_in[17]};
    const float* Wt[4] = {(const float*)d_in[6],  (const float*)d_in[10],
                          (const float*)d_in[14], (const float*)d_in[18]};
    const float* bt[4] = {(const float*)d_in[7],  (const float*)d_in[11],
                          (const float*)d_in[15], (const float*)d_in[19]};

    char* p = (char*)d_ws;
    auto alloc = [&](size_t bytes) -> char* {
        char* r = p;
        p += (bytes + 255) & ~(size_t)255;
        return r;
    };
    _Float16* wpk[4], *wtk[4];
    for (int i = 0; i < 4; ++i) wpk[i] = (_Float16*)alloc((size_t)HID * FAC * 2);
    for (int i = 0; i < 4; ++i) wtk[i] = (_Float16*)alloc((size_t)FAC * HID * 2);
    _Float16* xpack = (_Float16*)alloc((size_t)MROWS * HID * 2);
    _Float16* h16   = (_Float16*)alloc((size_t)MROWS * FAC * 2);
    _Float16* hpack = (_Float16*)alloc((size_t)MROWS * FAC * 2);
    _Float16* qh    = (_Float16*)alloc((size_t)BH * S_ * DK * 2);
    _Float16* kh    = (_Float16*)alloc((size_t)BH * S_ * DK * 2);
    _Float16* vh    = (_Float16*)alloc((size_t)BH * S_ * DK * 2);
    _Float16* qpack = (_Float16*)alloc((size_t)BH * S_ * DK * 2);
    _Float16* ktpack= (_Float16*)alloc((size_t)BH * S_ * DK * 2);
    _Float16* vpack = (_Float16*)alloc((size_t)BH * S_ * DK * 2);
    _Float16* cv    = qh;   // qh plain no longer needed once qpack exists

    // Pack all 8 weight matrices into B-fragment layout.
    for (int i = 0; i < 4; ++i) {
        hipLaunchKernelGGL(pack_b_kernel, dim3(128), dim3(128), 0, stream,
                           Wp[i], wpk[i], HID, FAC);
        hipLaunchKernelGGL(pack_b_kernel, dim3(128), dim3(128), 0, stream,
                           Wt[i], wtk[i], FAC, HID);
    }

    // Q / K / V factored projections -> head-split f16.
    const float* xin[3]  = {x_q, x_k, x_v};
    _Float16*    xout[3] = {qh, kh, vh};
    for (int t = 0; t < 3; ++t) {
        hipLaunchKernelGGL(pack_a_kernel<float>, dim3(4096), dim3(128), 0, stream,
                           xin[t], xpack, MROWS, HID);
        hipLaunchKernelGGL(gemm_wmma_kernel<0>, dim3(4, 64), dim3(128), 0, stream,
                           xpack, wpk[t], bp[t], (void*)h16, MROWS, FAC, HID);
        hipLaunchKernelGGL(pack_a_kernel<_Float16>, dim3(1024), dim3(128), 0, stream,
                           h16, hpack, MROWS, FAC);
        hipLaunchKernelGGL(gemm_wmma_kernel<1>, dim3(16, 64), dim3(128), 0, stream,
                           hpack, wtk[t], bt[t], (void*)xout[t], MROWS, HID, FAC);
    }

    // Pack attention operands into fragment layouts.
    hipLaunchKernelGGL(pack_a_kernel<_Float16>, dim3(4096), dim3(128), 0, stream,
                       qh, qpack, BH * S_, DK);
    hipLaunchKernelGGL(pack_kt_kernel, dim3(4096), dim3(128), 0, stream, kh, ktpack);
    hipLaunchKernelGGL(pack_v_kernel,  dim3(4096), dim3(128), 0, stream, vh, vpack);

    // Flash attention -> merged-head cv (f16, row-major [B*S, HID]).
    hipLaunchKernelGGL(flash_attn_kernel, dim3(32, 64), dim3(128), 0, stream,
                       qpack, ktpack, vpack, mask, cv);

    // Output factored projection -> fp32 d_out.
    hipLaunchKernelGGL(pack_a_kernel<_Float16>, dim3(4096), dim3(128), 0, stream,
                       cv, xpack, MROWS, HID);
    hipLaunchKernelGGL(gemm_wmma_kernel<0>, dim3(4, 64), dim3(128), 0, stream,
                       xpack, wpk[3], bp[3], (void*)h16, MROWS, FAC, HID);
    hipLaunchKernelGGL(pack_a_kernel<_Float16>, dim3(1024), dim3(128), 0, stream,
                       h16, hpack, MROWS, FAC);
    hipLaunchKernelGGL(gemm_wmma_kernel<2>, dim3(16, 64), dim3(128), 0, stream,
                       hpack, wtk[3], bt[3], d_out, MROWS, HID, FAC);
}
